// Attention_32495722562066
// MI455X (gfx1250) — compile-verified, matrix operand order checked by reading, not run on hardware
//
#include <hip/hip_runtime.h>
#include <hip/hip_bf16.h>

typedef __bf16 bhalf;
typedef __attribute__((ext_vector_type(16))) __bf16 v16bf;
typedef __attribute__((ext_vector_type(8)))  float  v8f;

#define WMMA_BF16(a, b, c) \
  __builtin_amdgcn_wmma_f32_16x16x32_bf16(false, (a), false, (b), (short)0, (c), false, false)

static __device__ inline bhalf f2bf(float f) {
  unsigned u = __builtin_bit_cast(unsigned, f);
  u += 0x7FFFu + ((u >> 16) & 1u);           // round-to-nearest-even
  return __builtin_bit_cast(bhalf, (unsigned short)(u >> 16));
}
static __device__ inline bhalf bf_lo(unsigned u) { return __builtin_bit_cast(bhalf, (unsigned short)(u & 0xFFFFu)); }
static __device__ inline bhalf bf_hi(unsigned u) { return __builtin_bit_cast(bhalf, (unsigned short)(u >> 16)); }

union Frag16 { v16bf v; float4 f[2]; };

// Build a 16xbf16 A/B fragment from two 16-byte chunks (global or LDS).
static __device__ inline v16bf make_frag(const bhalf* p0, const bhalf* p1) {
  Frag16 r;
  r.f[0] = *reinterpret_cast<const float4*>(p0);
  r.f[1] = *reinterpret_cast<const float4*>(p1);
  return r.v;
}

static constexpr int Bb = 2, T = 2048, C = 1024, H = 16, HD = 64;
static constexpr int M = Bb * T;  // 4096

// ---------------------------------------------------------------------------
// Tensor Data Mover: 2D tile load (global -> LDS), bf16 elements.
// D# layout per CDNA5 ISA. Hedged with
// __has_builtin; arity differs between clang-22 (5 args) and clang-23 (6).
// ---------------------------------------------------------------------------
#if defined(__has_builtin)
#if __has_builtin(__builtin_amdgcn_tensor_load_to_lds) && \
    __has_builtin(__builtin_amdgcn_s_wait_tensorcnt)
#define HAVE_TDM 1
#endif
#endif

#if defined(HAVE_TDM)
typedef unsigned int tdm_u32x4 __attribute__((ext_vector_type(4)));
typedef int          tdm_i32x8 __attribute__((ext_vector_type(8)));
typedef int          tdm_i32x4 __attribute__((ext_vector_type(4)));

static __device__ inline void tdm_load_2d_bf16(void* lds_dst, const bhalf* gsrc,
                                               int tile_cols, int tile_rows,
                                               int row_stride_elems) {
  const unsigned long long ga = (unsigned long long)(size_t)gsrc;
  const unsigned lds = (unsigned)(size_t)lds_dst;

  tdm_u32x4 g0;
  g0.x = 1u;                                         // count=1, user descriptor
  g0.y = lds;                                        // lds_addr
  g0.z = (unsigned)ga;                               // global_addr[31:0]
  g0.w = (unsigned)((ga >> 32) & 0x01FFFFFFu)        // global_addr[56:32]
       | (2u << 30);                                 // type = 2 ("image")

  const int td0 = 1 << 20, td1 = 1 << 20;            // large tensor dims (no OOB clip)
  const long long s0 = (long long)row_stride_elems;  // dim0 stride, data_size units
  tdm_i32x8 g1;
  g1[0] = 1 << 16;                                   // wg_mask=0, data_size=1 (2B)
  g1[1] = (int)((td0 & 0xFFFF) << 16);               // tensor_dim0[15:0]
  g1[2] = (int)(((td0 >> 16) & 0xFFFF) | ((td1 & 0xFFFF) << 16));
  g1[3] = (int)(((td1 >> 16) & 0xFFFF) | ((tile_cols & 0xFFFF) << 16));
  g1[4] = (int)(tile_rows & 0xFFFF);                 // tile_dim1; tile_dim2=0
  g1[5] = (int)(s0 & 0xFFFFFFFFll);                  // tensor_dim0_stride[31:0]
  g1[6] = (int)((s0 >> 32) & 0xFFFFll);              // stride[47:32]; dim1_stride=0
  g1[7] = 0;

  const tdm_i32x4 gz4 = {0, 0, 0, 0};                // 2D: groups 2/3 unused
#if __clang_major__ >= 23
  const tdm_i32x8 gz8 = {0, 0, 0, 0, 0, 0, 0, 0};
  __builtin_amdgcn_tensor_load_to_lds(g0, g1, gz4, gz4, gz8, 0);
#else
  __builtin_amdgcn_tensor_load_to_lds(g0, g1, gz4, gz4, 0);
#endif
}
#endif

// ---------------------------------------------------------------------------
// Kernel 1: fp32 -> bf16 conversion
// ---------------------------------------------------------------------------
__global__ void cvt_f32_bf16(const float* __restrict__ src, bhalf* __restrict__ dst, int n) {
  int i = blockIdx.x * blockDim.x + threadIdx.x;
  if (i < n) dst[i] = f2bf(src[i]);
}

// ---------------------------------------------------------------------------
// Kernel 2: QKV projection GEMM (y = x @ W^T) fused with per-head RMSNorm.
// grid = (M/128, H, 3); block = 128 (4 waves). Wave w computes a 32x64 tile
// (two 16-row subtiles sharing B fragments). Output head-major [B,H,T,64].
// ---------------------------------------------------------------------------
__global__ __launch_bounds__(128)
void qkv_gemm(const bhalf* __restrict__ xb,
              const bhalf* __restrict__ Wq, const bhalf* __restrict__ Wk,
              const bhalf* __restrict__ Wv,
              bhalf* __restrict__ Qo, bhalf* __restrict__ Ko, bhalf* __restrict__ Vo) {
  __shared__ bhalf Bs[64 * 32];              // W tile: 64 cols x 32 k (4 KB)

  const int tid  = threadIdx.x;
  const int lane = tid & 31;
  const int wave = tid >> 5;
  const int which = blockIdx.z;
  const bhalf* W   = (which == 0) ? Wq : (which == 1) ? Wk : Wv;
  bhalf*       Out = (which == 0) ? Qo : (which == 1) ? Ko : Vo;
  const bool do_rms = (which < 2);

  const int m_base = blockIdx.x * 128 + wave * 32;
  const int head   = blockIdx.y;
  const int n_base = head * 64;

  const v8f zero = {0.f,0.f,0.f,0.f,0.f,0.f,0.f,0.f};
  v8f acc[2][4] = {{zero, zero, zero, zero}, {zero, zero, zero, zero}};

  const int lhalf = (lane >> 4) & 1;         // 0: lanes 0-15, 1: lanes 16-31
  const int l15   = lane & 15;

  for (int k0 = 0; k0 < C; k0 += 32) {
    // Stage W[n_base + r][k0 .. k0+32) -> Bs.
#if defined(HAVE_TDM)
    if (tid == 0) {
      tdm_load_2d_bf16(Bs, W + (size_t)n_base * C + k0, /*cols*/32, /*rows*/64, /*stride*/C);
      __builtin_amdgcn_s_wait_tensorcnt((short)0);
    }
#else
    {
      const int r = tid >> 1;
      const int c = (tid & 1) * 16;
      const float4* src = reinterpret_cast<const float4*>(W + (size_t)(n_base + r) * C + k0 + c);
      float4* dst = reinterpret_cast<float4*>(&Bs[r * 32 + c]);
      dst[0] = src[0];
      dst[1] = src[1];
    }
#endif
    __syncthreads();

    // Two A fragments straight from global (L2 resident).
    v16bf af[2];
#pragma unroll
    for (int mt = 0; mt < 2; ++mt) {
      const bhalf* arow = xb + (size_t)(m_base + mt * 16 + l15) * C + k0 + lhalf * 8;
      __builtin_prefetch(arow + 32, 0, 3);   // global_prefetch_b8 for next K step
      af[mt] = make_frag(arow, arow + 16);
    }

#pragma unroll
    for (int nt = 0; nt < 4; ++nt) {
      const bhalf* brow = &Bs[(nt * 16 + l15) * 32 + lhalf * 8];
      v16bf bfrag = make_frag(brow, brow + 16);
#pragma unroll
      for (int mt = 0; mt < 2; ++mt)
        acc[mt][nt] = WMMA_BF16(af[mt], bfrag, acc[mt][nt]);
    }
    __syncthreads();
  }

  // In-register RMSNorm over the head dim (64 values per output row).
  if (do_rms) {
#pragma unroll
    for (int mt = 0; mt < 2; ++mt) {
#pragma unroll
      for (int j = 0; j < 8; ++j) {
        float ss = 0.f;
#pragma unroll
        for (int nt = 0; nt < 4; ++nt) ss += acc[mt][nt][j] * acc[mt][nt][j];
        ss += __shfl_xor(ss, 1, 32);
        ss += __shfl_xor(ss, 2, 32);
        ss += __shfl_xor(ss, 4, 32);
        ss += __shfl_xor(ss, 8, 32);
        const float scale = rsqrtf(ss * (1.0f / 64.0f) + 1.1920929e-7f);
#pragma unroll
        for (int nt = 0; nt < 4; ++nt) acc[mt][nt][j] *= scale;
      }
    }
  }

  // Store bf16, head-major [B, H, T, 64].
  const int rowoff = lhalf ? 8 : 0;
#pragma unroll
  for (int mt = 0; mt < 2; ++mt) {
#pragma unroll
    for (int j = 0; j < 8; ++j) {
      const int mg = m_base + mt * 16 + j + rowoff;
      const int bi = mg >> 11;               // / 2048
      const int t  = mg & 2047;
      const size_t base = (((size_t)bi * H + head) * T + t) * HD;
#pragma unroll
      for (int nt = 0; nt < 4; ++nt) {
        Out[base + nt * 16 + l15] = f2bf(acc[mt][nt][j]);
      }
    }
  }
}

// ---------------------------------------------------------------------------
// Kernel 3: flash attention. grid = (T/64, H, B); block = 128 (4 waves).
// Wave w owns 16 query rows. Key loop in 64-key blocks; K tile staged via
// TDM (if available), V staged transposed manually.
// ---------------------------------------------------------------------------
__global__ __launch_bounds__(128)
void attn_kernel(const bhalf* __restrict__ Qg, const bhalf* __restrict__ Kg,
                 const bhalf* __restrict__ Vg, const unsigned char* __restrict__ pmask,
                 bhalf* __restrict__ Yb) {
  __shared__ bhalf Ks[64 * 64];              // K block, [key][hd]           (8 KB)
  __shared__ bhalf Vts[64 * 64];             // V block transposed [hd][key] (8 KB)
  __shared__ bhalf Ps[4][16 * 64];           // per-wave P staging           (8 KB)

  const int tid  = threadIdx.x;
  const int lane = tid & 31;
  const int wave = tid >> 5;
  const int lhalf = (lane >> 4) & 1;
  const int l15   = lane & 15;

  const int b  = blockIdx.z;
  const int h  = blockIdx.y;
  const size_t bh = (size_t)b * H + h;
  const bhalf* Qb = Qg + bh * T * HD;
  const bhalf* Kb = Kg + bh * T * HD;
  const bhalf* Vb = Vg + bh * T * HD;
  const int q0 = blockIdx.x * 64 + wave * 16;

  // Q fragments for this wave's 16 rows (hd = 64 -> two 32-deep chunks).
  const bhalf* qrow = Qb + (size_t)(q0 + l15) * HD + lhalf * 8;
  const v16bf qf0 = make_frag(qrow,      qrow + 16);
  const v16bf qf1 = make_frag(qrow + 32, qrow + 48);

  const v8f zero = {0.f,0.f,0.f,0.f,0.f,0.f,0.f,0.f};
  v8f o[4] = {zero, zero, zero, zero};
  float mrow[8], lrow[8];
#pragma unroll
  for (int j = 0; j < 8; ++j) { mrow[j] = -3.0e38f; lrow[j] = 0.f; }

  for (int kb = 0; kb < T; kb += 64) {
    // --- stage K block ---
#if defined(HAVE_TDM)
    if (tid == 0) {
      tdm_load_2d_bf16(Ks, Kb + (size_t)kb * HD, /*cols*/HD, /*rows*/64, /*stride*/HD);
      __builtin_amdgcn_s_wait_tensorcnt((short)0);
    }
#else
    {
      const int r = tid >> 1;                // key row within block
      const int c = (tid & 1) * 32;          // hd offset
      const float4* ksrc = reinterpret_cast<const float4*>(Kb + (size_t)(kb + r) * HD + c);
      float4* kdst = reinterpret_cast<float4*>(&Ks[r * 64 + c]);
      kdst[0] = ksrc[0]; kdst[1] = ksrc[1]; kdst[2] = ksrc[2]; kdst[3] = ksrc[3];
    }
#endif
    // --- stage V block, transposed ---
    {
      const int r = tid >> 1;
      const int c = (tid & 1) * 32;
      const float4* vsrc = reinterpret_cast<const float4*>(Vb + (size_t)(kb + r) * HD + c);
#pragma unroll
      for (int q = 0; q < 4; ++q) {
        const float4 f = vsrc[q];
        const unsigned ux = __builtin_bit_cast(unsigned, f.x);
        const unsigned uy = __builtin_bit_cast(unsigned, f.y);
        const unsigned uz = __builtin_bit_cast(unsigned, f.z);
        const unsigned uw = __builtin_bit_cast(unsigned, f.w);
        const int d = c + q * 8;
        Vts[(d + 0) * 64 + r] = bf_lo(ux); Vts[(d + 1) * 64 + r] = bf_hi(ux);
        Vts[(d + 2) * 64 + r] = bf_lo(uy); Vts[(d + 3) * 64 + r] = bf_hi(uy);
        Vts[(d + 4) * 64 + r] = bf_lo(uz); Vts[(d + 5) * 64 + r] = bf_hi(uz);
        Vts[(d + 6) * 64 + r] = bf_lo(uw); Vts[(d + 7) * 64 + r] = bf_hi(uw);
      }
    }
    __syncthreads();

    // S = Q K^T (16 q x 64 keys per wave).
    v8f s[4] = {zero, zero, zero, zero};
#pragma unroll
    for (int nt = 0; nt < 4; ++nt) {
      const bhalf* brow = &Ks[(nt * 16 + l15) * 64 + lhalf * 8];
      v16bf b0 = make_frag(brow,      brow + 16);
      v16bf b1 = make_frag(brow + 32, brow + 48);
      s[nt] = WMMA_BF16(qf0, b0, s[nt]);
      s[nt] = WMMA_BF16(qf1, b1, s[nt]);
    }

    // Scale, mask, per-row block max.
    float tmax[8];
#pragma unroll
    for (int j = 0; j < 8; ++j) tmax[j] = -3.0e38f;
#pragma unroll
    for (int nt = 0; nt < 4; ++nt) {
      const int key = kb + nt * 16 + l15;
      const bool ok = pmask[(size_t)b * T + key] != 0;
#pragma unroll
      for (int j = 0; j < 8; ++j) {
        float v = s[nt][j] * 0.125f;         // 1/sqrt(64)
        v = ok ? v : -3.0e38f;
        s[nt][j] = v;
        tmax[j] = fmaxf(tmax[j], v);
      }
    }
#pragma unroll
    for (int j = 0; j < 8; ++j) {
      tmax[j] = fmaxf(tmax[j], __shfl_xor(tmax[j], 1, 32));
      tmax[j] = fmaxf(tmax[j], __shfl_xor(tmax[j], 2, 32));
      tmax[j] = fmaxf(tmax[j], __shfl_xor(tmax[j], 4, 32));
      tmax[j] = fmaxf(tmax[j], __shfl_xor(tmax[j], 8, 32));
    }

    // Online softmax rescale of running state.
    float alpha[8];
#pragma unroll
    for (int j = 0; j < 8; ++j) {
      const float mn = fmaxf(mrow[j], tmax[j]);
      alpha[j] = __expf(mrow[j] - mn);
      mrow[j]  = mn;
      lrow[j] *= alpha[j];
    }
#pragma unroll
    for (int nt = 0; nt < 4; ++nt)
#pragma unroll
      for (int j = 0; j < 8; ++j) o[nt][j] *= alpha[j];

    // P = exp(S - m); stage P (C-layout -> row-major LDS) for the PV WMMA.
    float psum[8];
#pragma unroll
    for (int j = 0; j < 8; ++j) psum[j] = 0.f;
    const int prow_off = lhalf ? 8 : 0;
#pragma unroll
    for (int nt = 0; nt < 4; ++nt) {
      const int col = nt * 16 + l15;
#pragma unroll
      for (int j = 0; j < 8; ++j) {
        const float p = __expf(s[nt][j] - mrow[j]);
        psum[j] += p;
        Ps[wave][(j + prow_off) * 64 + col] = f2bf(p);
      }
    }
#pragma unroll
    for (int j = 0; j < 8; ++j) {
      psum[j] += __shfl_xor(psum[j], 1, 32);
      psum[j] += __shfl_xor(psum[j], 2, 32);
      psum[j] += __shfl_xor(psum[j], 4, 32);
      psum[j] += __shfl_xor(psum[j], 8, 32);
      lrow[j] += psum[j];
    }
    __syncthreads();                          // P visible for A-fragment reads

    // O += P @ V  (A = P 16x64, B = V^T-staged 64x64).
    const bhalf* prow = &Ps[wave][l15 * 64 + lhalf * 8];
    v16bf p0 = make_frag(prow,      prow + 16);
    v16bf p1 = make_frag(prow + 32, prow + 48);
#pragma unroll
    for (int nt = 0; nt < 4; ++nt) {
      const bhalf* vrow = &Vts[(nt * 16 + l15) * 64 + lhalf * 8];
      v16bf vb0 = make_frag(vrow,      vrow + 16);
      v16bf vb1 = make_frag(vrow + 32, vrow + 48);
      o[nt] = WMMA_BF16(p0, vb0, o[nt]);
      o[nt] = WMMA_BF16(p1, vb1, o[nt]);
    }
    __syncthreads();                          // done with Ks/Vts before next block
  }

  // Normalize and store Y (row-major [B*T, C], col = h*64 + d) as bf16.
  const int rowoff = lhalf ? 8 : 0;
#pragma unroll
  for (int j = 0; j < 8; ++j) {
    const float inv = (lrow[j] > 0.f) ? (1.f / lrow[j]) : 0.f;
    const int t = q0 + j + rowoff;
    const size_t row = (size_t)b * T + t;
#pragma unroll
    for (int nt = 0; nt < 4; ++nt) {
      Yb[row * C + h * 64 + nt * 16 + l15] = f2bf(o[nt][j] * inv);
    }
  }
}

// ---------------------------------------------------------------------------
// Kernel 4: output projection out = Y @ Wo^T, fp32 output.
// grid = (M/128, C/64); block = 128, 32x64 wave tiles.
// ---------------------------------------------------------------------------
__global__ __launch_bounds__(128)
void out_gemm(const bhalf* __restrict__ Yb, const bhalf* __restrict__ Wo,
              float* __restrict__ out) {
  __shared__ bhalf Bs[64 * 32];

  const int tid  = threadIdx.x;
  const int lane = tid & 31;
  const int wave = tid >> 5;
  const int lhalf = (lane >> 4) & 1;
  const int l15   = lane & 15;

  const int m_base = blockIdx.x * 128 + wave * 32;
  const int n_base = blockIdx.y * 64;

  const v8f zero = {0.f,0.f,0.f,0.f,0.f,0.f,0.f,0.f};
  v8f acc[2][4] = {{zero, zero, zero, zero}, {zero, zero, zero, zero}};

  for (int k0 = 0; k0 < C; k0 += 32) {
#if defined(HAVE_TDM)
    if (tid == 0) {
      tdm_load_2d_bf16(Bs, Wo + (size_t)n_base * C + k0, /*cols*/32, /*rows*/64, /*stride*/C);
      __builtin_amdgcn_s_wait_tensorcnt((short)0);
    }
#else
    {
      const int r = tid >> 1;
      const int c = (tid & 1) * 16;
      const float4* src = reinterpret_cast<const float4*>(Wo + (size_t)(n_base + r) * C + k0 + c);
      float4* dst = reinterpret_cast<float4*>(&Bs[r * 32 + c]);
      dst[0] = src[0];
      dst[1] = src[1];
    }
#endif
    __syncthreads();

    v16bf af[2];
#pragma unroll
    for (int mt = 0; mt < 2; ++mt) {
      const bhalf* arow = Yb + (size_t)(m_base + mt * 16 + l15) * C + k0 + lhalf * 8;
      __builtin_prefetch(arow + 32, 0, 3);
      af[mt] = make_frag(arow, arow + 16);
    }

#pragma unroll
    for (int nt = 0; nt < 4; ++nt) {
      const bhalf* brow = &Bs[(nt * 16 + l15) * 32 + lhalf * 8];
      v16bf bfrag = make_frag(brow, brow + 16);
#pragma unroll
      for (int mt = 0; mt < 2; ++mt)
        acc[mt][nt] = WMMA_BF16(af[mt], bfrag, acc[mt][nt]);
    }
    __syncthreads();
  }

  const int rowoff = lhalf ? 8 : 0;
#pragma unroll
  for (int mt = 0; mt < 2; ++mt) {
#pragma unroll
    for (int j = 0; j < 8; ++j) {
      const size_t mg = (size_t)(m_base + mt * 16 + j + rowoff);
#pragma unroll
      for (int nt = 0; nt < 4; ++nt) {
        out[mg * C + n_base + nt * 16 + l15] = acc[mt][nt][j];
      }
    }
  }
}

// ---------------------------------------------------------------------------
// Host-side launcher
// ---------------------------------------------------------------------------
extern "C" void kernel_launch(void* const* d_in, const int* in_sizes, int n_in,
                              void* d_out, int out_size, void* d_ws, size_t ws_size,
                              hipStream_t stream) {
  (void)in_sizes; (void)n_in; (void)out_size; (void)ws_size;

  const float* x  = (const float*)d_in[0];
  const unsigned char* pm = (const unsigned char*)d_in[1];
  const float* Wq = (const float*)d_in[2];
  const float* Wk = (const float*)d_in[3];
  const float* Wv = (const float*)d_in[4];
  const float* Wo = (const float*)d_in[5];
  float* out = (float*)d_out;

  bhalf* ws  = (bhalf*)d_ws;
  bhalf* xb  = ws;                            // M*C
  bhalf* Wqb = xb  + (size_t)M * C;           // C*C
  bhalf* Wkb = Wqb + (size_t)C * C;
  bhalf* Wvb = Wkb + (size_t)C * C;
  bhalf* Wob = Wvb + (size_t)C * C;
  bhalf* Qb  = Wob + (size_t)C * C;           // [B,H,T,64] = M*C
  bhalf* Kb  = Qb  + (size_t)M * C;
  bhalf* Vb  = Kb  + (size_t)M * C;
  bhalf* Yb  = Vb  + (size_t)M * C;           // [M, C]

  const int nX = M * C, nW = C * C;
  cvt_f32_bf16<<<(nX + 255) / 256, 256, 0, stream>>>(x,  xb,  nX);
  cvt_f32_bf16<<<(nW + 255) / 256, 256, 0, stream>>>(Wq, Wqb, nW);
  cvt_f32_bf16<<<(nW + 255) / 256, 256, 0, stream>>>(Wk, Wkb, nW);
  cvt_f32_bf16<<<(nW + 255) / 256, 256, 0, stream>>>(Wv, Wvb, nW);
  cvt_f32_bf16<<<(nW + 255) / 256, 256, 0, stream>>>(Wo, Wob, nW);

  dim3 gQKV(M / 128, H, 3);
  qkv_gemm<<<gQKV, 128, 0, stream>>>(xb, Wqb, Wkb, Wvb, Qb, Kb, Vb);

  dim3 gAttn(T / 64, H, Bb);
  attn_kernel<<<gAttn, 128, 0, stream>>>(Qb, Kb, Vb, pm, Yb);

  dim3 gOut(M / 128, C / 64, 1);
  out_gemm<<<gOut, 128, 0, stream>>>(Yb, Wob, out);
}
